// MemoryRel_46978352284448
// MI455X (gfx1250) — matrix operand order repaired and checked by reading, not compiled, not run on hardware
//
#include <hip/hip_runtime.h>

#define LL 147456      // 384*384
#define LDIM 384
#define DDIM 512

typedef __attribute__((ext_vector_type(16))) __bf16 v16bf;
typedef __attribute__((ext_vector_type(8)))  float  v8f;

__device__ __forceinline__ float bf2f(unsigned short u) {
  return __uint_as_float(((unsigned)u) << 16);
}
__device__ __forceinline__ float bflo(unsigned d) { return __uint_as_float(d << 16); }
__device__ __forceinline__ float bfhi(unsigned d) { return __uint_as_float(d & 0xFFFF0000u); }
__device__ __forceinline__ unsigned short f2bf(float f) {
  unsigned u = __float_as_uint(f);
  u = (u + 0x7FFFu + ((u >> 16) & 1u)) >> 16;
  return (unsigned short)u;
}
__device__ __forceinline__ float leaky(float v) { return v > 0.f ? v : 0.01f * v; }

// ---------------- phase kernels ----------------

// init inputs = concat(e1,e2); zero mask flags
__global__ void k_init(const float* e1, const float* e2, float* inp, int* flags) {
  int t = blockIdx.x * 256 + threadIdx.x;          // grid 4 x 256
  inp[t] = (t < 512) ? e1[t] : e2[t - 512];
  if (t < 3) flags[t] = 0;
}

// marg = sum_r E[r, idx]; any-flags for mask fallback thresholds
__global__ void k_marg(const float* E, float* marg, int* flags) {
  __shared__ int lf[3];
  int t = threadIdx.x;
  if (t < 3) lf[t] = 0;
  __syncthreads();
  int idx = blockIdx.x * 256 + t;                  // grid 576 x 256 == LL
  float acc = 0.f;
  for (int r = 0; r < 45; ++r) acc += E[(size_t)r * LL + idx];
  marg[idx] = acc;
  if      (acc > 1e-6f) atomicOr(&lf[0], 1);
  else if (acc > 1e-7f) atomicOr(&lf[1], 1);
  else if (acc > 1e-8f) atomicOr(&lf[2], 1);
  __syncthreads();
  if (t < 3 && lf[t]) atomicOr(&flags[t], 1);
}

// T = rel_embs @ Wc[512:527,:]  stored transposed + zero padded to K=64, bf16:
// Tt[n*64 + r], r in [0,64)
__global__ void k_T(const float* rel, const float* Wc, unsigned short* Tt) {
  int r = blockIdx.x;          // 0..63
  int n = threadIdx.x;         // 0..511
  float acc = 0.f;
  if (r < 45) {
    for (int e = 0; e < 15; ++e)
      acc += rel[r * 15 + e] * Wc[(size_t)(512 + e) * DDIM + n];
  }
  Tt[(size_t)n * 64 + r] = f2bf(acc);
}

// P = H @ Wc[0:512,:],  Q = H @ Wc[527:1039,:]
__global__ void k_PQ(const float* H, const float* Wc, float* P, float* Q) {
  __shared__ float hrow[512];
  int t = threadIdx.x;
  int i = blockIdx.x >> 1;                         // grid (768, 2) x 256
  int n = ((blockIdx.x & 1) << 8) + t;
  int mat = blockIdx.y;
  hrow[t]       = H[(size_t)i * DDIM + t];
  hrow[t + 256] = H[(size_t)i * DDIM + t + 256];
  __syncthreads();
  size_t rowoff = mat ? 527 : 0;
  float acc = 0.f;
  for (int k = 0; k < 512; ++k) acc += hrow[k] * Wc[(rowoff + k) * DDIM + n];
  (mat ? Q : P)[(size_t)i * DDIM + n] = acc;
}

// mem_bank[idx,n] = leaky( w*(P[i,n]+Q[j,n]) + (E^T @ T)[idx,n] + bc[n] ), bf16 out.
// One wave per 16-row idx tile; S-term via two chained v_wmma_f32_16x16x32_bf16.
// Tt (64KB) staged block-wide into LDS via GLOBAL_LOAD_ASYNC_TO_LDS_B128.
__global__ void k_membank(const float* __restrict__ E, const unsigned short* __restrict__ Tt,
                          const float* __restrict__ P, const float* __restrict__ Q,
                          const float* __restrict__ marg, const float* __restrict__ bc,
                          unsigned short* __restrict__ mb) {
  __shared__ __align__(32) unsigned short tlds[64 * 512];   // 64 KB
  int t = threadIdx.x;
  int lane = t & 31;

  // cooperative async copy: each thread moves 16 x 16B = 256B (ASYNCcnt-tracked)
  {
    unsigned lds0 = (unsigned)(uintptr_t)(void*)tlds;       // low 32 bits = LDS offset
#pragma unroll
    for (int it = 0; it < 16; ++it) {
      int eofs = (t + it * 256) * 8;                        // in ushorts (8 = 16B)
      unsigned      ldsa = lds0 + (unsigned)eofs * 2u;
      unsigned long long ga = (unsigned long long)(uintptr_t)(Tt + eofs);
      asm volatile("global_load_async_to_lds_b128 %0, %1, off"
                   :: "v"(ldsa), "v"(ga) : "memory");
    }
    asm volatile("s_wait_asynccnt 0x0" ::: "memory");
  }
  __syncthreads();

  int gw = blockIdx.x * 8 + (t >> 5);              // grid 1152 x 256 -> 9216 waves
  int idx0 = gw << 4;                              // 16 rows per wave
  int i  = idx0 / LDIM;
  int j0 = idx0 % LDIM;                            // LDIM % 16 == 0 -> single i per tile
  int m = lane & 15;
  bool hi = lane >= 16;

  // A fragment: contiguous k order used by this code (same as B):
  // lane<16 holds M=lane, K in {kb..kb+15}, kb=0; lane>=16: kb=16.
  v16bf a0, a1;
  {
    int kb = hi ? 16 : 0;
#pragma unroll
    for (int e = 0; e < 16; ++e) {
      int r0 = kb + e;                             // 0..31, always < 45
      float f0 = E[(size_t)r0 * LL + idx0 + m];
      int r1 = r0 + 32;                            // 32..63, pad with 0 past 44
      float f1 = (r1 < 45) ? E[(size_t)r1 * LL + idx0 + m] : 0.f;
      a0[e] = (__bf16)f0;
      a1[e] = (__bf16)f1;
    }
  }

  float wv[8];
#pragma unroll
  for (int g = 0; g < 8; ++g) wv[g] = marg[idx0 + g + (hi ? 8 : 0)];

  int N   = lane & 15;
  int kbb = hi ? 16 : 0;    // B 32x16 layout: lane<16 -> K base 0, lane>=16 -> K base 16

  for (int nt = 0; nt < 32; ++nt) {
    int n0 = nt << 4;
    const __bf16* tp = (const __bf16*)tlds + (size_t)(n0 + N) * 64 + kbb;
    v16bf b0 = *(const v16bf*)tp;          // ds_load: K = kbb..kbb+15   (rows 0..31)
    v16bf b1 = *(const v16bf*)(tp + 32);   // ds_load: K = 32+kbb..       (rows 32..63)
    v8f c = {};
    c = __builtin_amdgcn_wmma_f32_16x16x32_bf16(false, a0, false, b0, (short)0, c, false, false);
    c = __builtin_amdgcn_wmma_f32_16x16x32_bf16(false, a1, false, b1, (short)0, c, false, false);

    int n = n0 + N;
    float pv  = P[(size_t)i * DDIM + n];
    float bcv = bc[n];
#pragma unroll
    for (int g = 0; g < 8; ++g) {
      int M = g + (hi ? 8 : 0);                    // C/D layout: vgpr g, lanes 16-31 -> M=g+8
      float qv = Q[(size_t)(j0 + M) * DDIM + n];
      float val = leaky(wv[g] * (pv + qv) + c[g] + bcv);
      mb[(size_t)(idx0 + M) * DDIM + n] = f2bf(val);
    }
  }
}

// ---------------- hop kernels ----------------

// kv = inputs @ Wk[hop] + bk[hop]; k_ = tanh(kv[:512]); v_ = leaky(kv[512:])
__global__ void k_kv(const float* inp, const float* Wk, const float* bk, int hop,
                     float* kbuf, float* vbuf) {
  __shared__ float xin[1024];
  int t = threadIdx.x;
  int n = blockIdx.x * 256 + t;                    // grid 4 x 256
  for (int k = t; k < 1024; k += 256) xin[k] = inp[k];
  __syncthreads();
  float acc = bk[hop * 1024 + n];
  const float* wcol = Wk + (size_t)hop * 1024 * 1024 + n;
  for (int k = 0; k < 1024; ++k) acc += xin[k] * wcol[(size_t)k * 1024];
  if (n < 512) kbuf[n] = tanhf(acc);
  else         vbuf[n - 512] = leaky(acc);
}

// scores[idx] = maskf ? dot(mem_bank[idx], k_) : -1e30 ; per-block max partial
__global__ void k_scores(const unsigned short* __restrict__ mb, const float* __restrict__ kvec,
                         const float* __restrict__ marg, const int* __restrict__ flags,
                         float* __restrict__ scores, float* __restrict__ pmax) {
  __shared__ float kl[512];
  __shared__ float wmax[8];
  int t = threadIdx.x;
  kl[t] = kvec[t]; kl[t + 256] = kvec[t + 256];
  __syncthreads();
  float thresh = flags[0] ? 1e-6f : (flags[1] ? 1e-7f : 1e-8f);
  int lane = t & 31;
  int wid = t >> 5;
  int gw = blockIdx.x * 8 + wid;                   // grid 1152 x 256 -> 9216 waves
  float lmax = -3e38f;
  for (int it = 0; it < 16; ++it) {
    int row = gw + 9216 * it;
    const unsigned short* rp = mb + (size_t)row * DDIM;
    float acc = 0.f;
#pragma unroll
    for (int q = 0; q < 2; ++q) {
      int base = lane * 8 + q * 256;               // 8 bf16 = 16B per lane
      uint4 d = *(const uint4*)(rp + base);
      acc += bflo(d.x) * kl[base + 0] + bfhi(d.x) * kl[base + 1]
           + bflo(d.y) * kl[base + 2] + bfhi(d.y) * kl[base + 3]
           + bflo(d.z) * kl[base + 4] + bfhi(d.z) * kl[base + 5]
           + bflo(d.w) * kl[base + 6] + bfhi(d.w) * kl[base + 7];
    }
#pragma unroll
    for (int off = 16; off > 0; off >>= 1) acc += __shfl_down(acc, off, 32);
    if (lane == 0) {
      float sc = (marg[row] > thresh) ? acc : -1e30f;
      scores[row] = sc;
      lmax = fmaxf(lmax, sc);
    }
  }
  if (lane == 0) wmax[wid] = lmax;
  __syncthreads();
  if (t == 0) {
    float mx = wmax[0];
    for (int k = 1; k < 8; ++k) mx = fmaxf(mx, wmax[k]);
    pmax[blockIdx.x] = mx;
  }
}

__global__ void k_rmax(const float* pmax, float* gstats, int n) {
  __shared__ float red[256];
  int t = threadIdx.x;
  float mx = -3e38f;
  for (int k = t; k < n; k += 256) mx = fmaxf(mx, pmax[k]);
  red[t] = mx; __syncthreads();
  for (int s = 128; s; s >>= 1) { if (t < s) red[t] = fmaxf(red[t], red[t + s]); __syncthreads(); }
  if (t == 0) gstats[0] = red[0];
}

// scores <- exp(scores - gmax); per-block sum partial
__global__ void k_exp(float* scores, const float* gstats, float* psum) {
  __shared__ float red[256];
  int t = threadIdx.x;
  int idx = blockIdx.x * 256 + t;                  // grid 576 x 256 == LL
  float e = __expf(scores[idx] - gstats[0]);
  scores[idx] = e;
  red[t] = e; __syncthreads();
  for (int s = 128; s; s >>= 1) { if (t < s) red[t] += red[t + s]; __syncthreads(); }
  if (t == 0) psum[blockIdx.x] = red[0];
}

__global__ void k_rsum(const float* psum, float* gstats, int n) {
  __shared__ float red[256];
  int t = threadIdx.x;
  float s = 0.f;
  for (int k = t; k < n; k += 256) s += psum[k];
  red[t] = s; __syncthreads();
  for (int st = 128; st; st >>= 1) { if (t < st) red[t] += red[t + st]; __syncthreads(); }
  if (t == 0) gstats[1] = 1.0f / red[0];
}

// partial mem_rep: block b owns rows [b*256, b*256+256); each thread owns cols {2t, 2t+1}
__global__ void k_mrep_part(const unsigned short* __restrict__ mb,
                            const float* __restrict__ scores, float* __restrict__ part) {
  int t = threadIdx.x;
  int b = blockIdx.x;                              // grid 576 x 256
  size_t row0 = (size_t)b * 256;
  float a0 = 0.f, a1 = 0.f;
  for (int r = 0; r < 256; ++r) {
    size_t row = row0 + r;
    float e = scores[row];
    unsigned d = *(const unsigned*)(mb + row * DDIM + 2 * t);   // 2 bf16, b32 load
    a0 += e * bflo(d);
    a1 += e * bfhi(d);
  }
  float2* pp = (float2*)(part + (size_t)b * DDIM + 2 * t);
  *pp = make_float2(a0, a1);
}

__global__ void k_mrep_final(const float* part, const float* gstats, float* mrep) {
  int n = threadIdx.x;                             // 1 x 512
  float acc = 0.f;
  for (int b = 0; b < 576; ++b) acc += part[(size_t)b * DDIM + n];
  mrep[n] = acc * gstats[1];
}

// inputs_new = leaky(concat(v_, mem_rep) @ Wh[hop] + bh[hop])
__global__ void k_update(const float* vbuf, const float* mrep, const float* Wh,
                         const float* bh, int hop, float* out) {
  __shared__ float xin[1024];
  int t = threadIdx.x;
  int n = blockIdx.x * 256 + t;                    // grid 4 x 256
  for (int k = t; k < 1024; k += 256) xin[k] = (k < 512) ? vbuf[k] : mrep[k - 512];
  __syncthreads();
  float acc = bh[hop * 1024 + n];
  const float* wcol = Wh + (size_t)hop * 1024 * 1024 + n;
  for (int k = 0; k < 1024; ++k) acc += xin[k] * wcol[(size_t)k * 1024];
  out[n] = leaky(acc);
}

// ---------------- launch ----------------

extern "C" void kernel_launch(void* const* d_in, const int* in_sizes, int n_in,
                              void* d_out, int out_size, void* d_ws, size_t ws_size,
                              hipStream_t stream) {
  (void)in_sizes; (void)n_in; (void)out_size; (void)ws_size;
  const float* E    = (const float*)d_in[0];   // (1,45,384,384)
  const float* H    = (const float*)d_in[1];   // (1,384,512)
  const float* e1   = (const float*)d_in[2];
  const float* e2   = (const float*)d_in[3];
  // d_in[4] = sent_len (int, ==384) ignored
  const float* rel  = (const float*)d_in[5];   // (45,15)
  const float* Wc   = (const float*)d_in[6];   // (1039,512)
  const float* bc   = (const float*)d_in[7];   // (512,)
  const float* Wk   = (const float*)d_in[8];   // (3,1024,1024)
  const float* bk   = (const float*)d_in[9];   // (3,1024)
  const float* Wh   = (const float*)d_in[10];  // (3,1024,1024)
  const float* bh   = (const float*)d_in[11];  // (3,1024)

  char* ws = (char*)d_ws;
  // all offsets 256B-aligned
  constexpr size_t O_MARG  = 0;                         // LL f32      = 589824
  constexpr size_t O_FLAGS = 589824;                    // 3 int       (256)
  constexpr size_t O_GST   = 590080;                    // gmax, invZ  (256)
  constexpr size_t O_TT    = 590336;                    // 512*64 bf16 = 65536
  constexpr size_t O_P     = 655872;                    // 384*512 f32 = 786432
  constexpr size_t O_Q     = 1442304;                   // 786432
  constexpr size_t O_INP   = 2228736;                   // 1024 f32
  constexpr size_t O_K     = 2232832;                   // 512 f32
  constexpr size_t O_V     = 2234880;                   // 512 f32
  constexpr size_t O_SC    = 2236928;                   // LL f32      = 589824
  constexpr size_t O_PMAX  = 2826752;                   // 1152 f32 (pad 8192)
  constexpr size_t O_PSUM  = 2834944;                   // 576 f32  (pad 4096)
  constexpr size_t O_MRPP  = 2839040;                   // 576*512 f32 = 1179648
  constexpr size_t O_MREP  = 4018688;                   // 512 f32 (pad 2048)
  constexpr size_t O_MB    = 4020736;                   // LL*512 bf16 = 150994944

  float*          marg  = (float*)(ws + O_MARG);
  int*            flags = (int*)(ws + O_FLAGS);
  float*          gst   = (float*)(ws + O_GST);
  unsigned short* Tt    = (unsigned short*)(ws + O_TT);
  float*          P     = (float*)(ws + O_P);
  float*          Q     = (float*)(ws + O_Q);
  float*          inp   = (float*)(ws + O_INP);
  float*          kbuf  = (float*)(ws + O_K);
  float*          vbuf  = (float*)(ws + O_V);
  float*          sc    = (float*)(ws + O_SC);
  float*          pmax  = (float*)(ws + O_PMAX);
  float*          psum  = (float*)(ws + O_PSUM);
  float*          mrpp  = (float*)(ws + O_MRPP);
  float*          mrep  = (float*)(ws + O_MREP);
  unsigned short* mb    = (unsigned short*)(ws + O_MB);

  k_init<<<4, 256, 0, stream>>>(e1, e2, inp, flags);
  k_marg<<<576, 256, 0, stream>>>(E, marg, flags);
  k_T<<<64, 512, 0, stream>>>(rel, Wc, Tt);
  k_PQ<<<dim3(768, 2), 256, 0, stream>>>(H, Wc, P, Q);
  k_membank<<<1152, 256, 0, stream>>>(E, Tt, P, Q, marg, bc, mb);

  for (int hop = 0; hop < 3; ++hop) {
    k_kv<<<4, 256, 0, stream>>>(inp, Wk, bk, hop, kbuf, vbuf);
    k_scores<<<1152, 256, 0, stream>>>(mb, kbuf, marg, flags, sc, pmax);
    k_rmax<<<1, 256, 0, stream>>>(pmax, gst, 1152);
    k_exp<<<576, 256, 0, stream>>>(sc, gst, psum);
    k_rsum<<<1, 256, 0, stream>>>(psum, gst, 576);
    k_mrep_part<<<576, 256, 0, stream>>>(mb, sc, mrpp);
    k_mrep_final<<<1, 512, 0, stream>>>(mrpp, gst, mrep);
    float* outp = (hop == 2) ? (float*)d_out : inp;
    k_update<<<4, 256, 0, stream>>>(vbuf, mrep, Wh, bh, hop, outp);
  }
}
